// PCNNEncoder_44555990729450
// MI455X (gfx1250) — matrix operand order repaired and checked by reading, not compiled, not run on hardware
//
#include <hip/hip_runtime.h>
#include <hip/hip_bf16.h>
#include <stdint.h>

typedef __attribute__((ext_vector_type(16))) _Float16 v16h;
typedef __attribute__((ext_vector_type(8)))  float    v8f;

#define B_    64
#define L_    512
#define E_    128
#define C_    256
#define LPAD  518        // 3 (max lo) + 512 + 3 (max hi)
#define CT    768
#define OUTS  2304       // 3 * 768
#define XROWS 38         // rows staged per block: covers all shifts for Ks<=7

#if __has_builtin(__builtin_amdgcn_global_load_async_to_lds_b128)
#define USE_ASYNC_LDS 1
// Signature (from compiler diagnostic): param1 = AS(1) pointer to 16B int
// vector (global source), param2 = LDS dest, then imm offset + imm cpol.
typedef int v4i_raw __attribute__((vector_size(16)));
typedef __attribute__((address_space(1))) v4i_raw* as1_v4i;
typedef __attribute__((address_space(3))) v4i_raw* as3_v4i;
#else
#define USE_ASYNC_LDS 0
#endif

// ---------------- prep: f32 -> f16 convert (conv weights) ----------------
__global__ void cvt_f32_f16(const float* __restrict__ src, _Float16* __restrict__ dst, int n) {
    int i = blockIdx.x * blockDim.x + threadIdx.x;
    if (i < n) dst[i] = (_Float16)src[i];
}

// ---------------- prep: embedding gather to padded f16 (B, LPAD, E) -------
__global__ void gather_embed(const int* __restrict__ tok, const float* __restrict__ emb,
                             _Float16* __restrict__ xpad) {
    int row = blockIdx.x;                 // b*LPAD + lp
    int e   = threadIdx.x;                // 0..127
    int b   = row / LPAD, lp = row % LPAD;
    _Float16 v = (_Float16)0.0f;
    if (lp >= 3 && lp < 3 + L_) {
        int t = tok[b * L_ + (lp - 3)];
        v = (_Float16)emb[(size_t)t * E_ + e];
    }
    xpad[(size_t)row * E_ + e] = v;
}

// ---------------- prep: first occurrence of tokens 1,2,3 per row ----------
__global__ void seg_bounds(const int* __restrict__ tok, int* __restrict__ bounds) {
    int b = blockIdx.x * blockDim.x + threadIdx.x;
    if (b >= B_) return;
    int f0 = -1, f1 = -1, f2 = -1;
    for (int l = 0; l < L_; ++l) {
        int t = tok[b * L_ + l];
        if (t == 1 && f0 < 0) f0 = l;
        if (t == 2 && f1 < 0) f1 = l;
        if (t == 3 && f2 < 0) f2 = l;
    }
    if (f0 < 0) f0 = 0;                    // jnp.argmax default when absent
    if (f1 < 0) f1 = 0;
    if (f2 < 0) f2 = 0;
    bounds[b * 6 + 0] = 0;  bounds[b * 6 + 1] = f0;   // segment 0: [0,  f0]
    bounds[b * 6 + 2] = f0; bounds[b * 6 + 3] = f1;   // segment 1: [f0, f1]
    bounds[b * 6 + 4] = f1; bounds[b * 6 + 5] = f2;   // segment 2: [f1, f2]
}

// ---------------- prep: zero output (safe floor for ReLU'd maxima) --------
__global__ void zero_out(float* __restrict__ out, int n) {
    int i = blockIdx.x * blockDim.x + threadIdx.x;
    if (i < n) out[i] = 0.0f;
}

// ---- fused conv (WMMA f16->f32) + bias + ReLU + segment max, per Ks ------
// grid: (L_/32, 4, B_), block 128 (4 waves). Block stages a 38x128 f16 slab
// of padded activations into LDS once (async copy, ASYNCcnt); all 4 waves
// read B fragments from LDS (4x reuse). Each wave: 16 channels x 32
// positions = two 16x16 D tiles sharing every A (weight) fragment.
template<int Ks, int LO, int KSG>
__global__ __launch_bounds__(128) void conv_wmma_t(
    const _Float16* __restrict__ xpad,
    const _Float16* __restrict__ W,
    const float* __restrict__ bias,
    const int* __restrict__ bounds, float* __restrict__ out)
{
    __shared__ __align__(16) _Float16 xs[XROWS * E_];   // 9728 B

    const int lane = threadIdx.x & 31;
    const int wave = threadIdx.x >> 5;        // 0..3
    const int hh   = lane >> 4;               // half-wave: 0 or 1
    const int m    = lane & 15;               // A-row / B-col within half

    const int bb = blockIdx.z;
    const int l0 = blockIdx.x * 32;
    const int c0 = (blockIdx.y * 4 + wave) * 16;

    // ---- stage padded rows [l0 .. l0+37] into LDS ------------------------
    {
        const char* gsrc = (const char*)(xpad + ((size_t)bb * LPAD + l0) * E_);
        char*       ldst = (char*)xs;
        constexpr int NCH = XROWS * E_ * 2 / 16;          // 608 x 16B chunks
        for (int i = threadIdx.x; i < NCH; i += 128) {
#if USE_ASYNC_LDS
            __builtin_amdgcn_global_load_async_to_lds_b128(
                (as1_v4i)(gsrc + i * 16),
                (as3_v4i)(ldst + i * 16),
                0, 0);
#else
            *(int4*)(ldst + i * 16) = *(const int4*)(gsrc + i * 16);
#endif
        }
#if USE_ASYNC_LDS
#if __has_builtin(__builtin_amdgcn_s_wait_asynccnt)
        __builtin_amdgcn_s_wait_asynccnt(0);
#else
        asm volatile("s_wait_asynccnt 0" ::: "memory");
#endif
#endif
        __syncthreads();
    }

    v8f acc0 = {}, acc1 = {};

    #pragma unroll
    for (int koff = 0; koff < Ks; ++koff) {
        // A: per-lane weight row; B: LDS row = m + koff - LO + 3 (+16 for tile1)
        const _Float16* wrow = W + ((size_t)(c0 + m) * Ks + koff) * E_ + hh * 16;
        const _Float16* xrow = xs + (m + koff - LO + 3) * E_ + hh * 16;
        #pragma unroll
        for (int ec = 0; ec < 4; ++ec) {
            const int e0 = ec * 32;
            v16h a, bf0, bf1;
            // A fragment (16x32): half0 K=0..15 @e0;
            //                     half1 K=16..31 @e0+16
            int4* a4 = reinterpret_cast<int4*>(&a);
            a4[0] = *reinterpret_cast<const int4*>(wrow + e0);
            a4[1] = *reinterpret_cast<const int4*>(wrow + e0 + 8);
            // B fragments (32x16): K contiguous along E, from LDS.
            int4* b04 = reinterpret_cast<int4*>(&bf0);
            b04[0] = *reinterpret_cast<const int4*>(xrow + e0);
            b04[1] = *reinterpret_cast<const int4*>(xrow + e0 + 8);
            int4* b14 = reinterpret_cast<int4*>(&bf1);
            b14[0] = *reinterpret_cast<const int4*>(xrow + 16 * E_ + e0);
            b14[1] = *reinterpret_cast<const int4*>(xrow + 16 * E_ + e0 + 8);

            acc0 = __builtin_amdgcn_wmma_f32_16x16x32_f16(
                false, a, false, bf0, (short)0, acc0, false, false);
            acc1 = __builtin_amdgcn_wmma_f32_16x16x32_f16(
                false, a, false, bf1, (short)0, acc1, false, false);
        }
    }

    // Epilogue: bias + ReLU in registers.
    // C/D layout: VGPR r of lane holds (M = r + 8*hh, N = m).
    float val0[8], val1[8];
    #pragma unroll
    for (int r = 0; r < 8; ++r) {
        const float bv = bias[c0 + r + hh * 8];
        float v0 = acc0[r] + bv;
        float v1 = acc1[r] + bv;
        val0[r] = v0 > 0.0f ? v0 : 0.0f;
        val1[r] = v1 > 0.0f ? v1 : 0.0f;
    }

    // Segment bounds (uniform per block): load once.
    int pv[3], id[3];
    #pragma unroll
    for (int t = 0; t < 3; ++t) {
        pv[t] = bounds[bb * 6 + t * 2];
        id[t] = bounds[bb * 6 + t * 2 + 1];
    }

    const int cgb = KSG * C_ + c0;           // global channel base
    #pragma unroll
    for (int nt = 0; nt < 2; ++nt) {
        const int lt = l0 + nt * 16;         // this tile's first position
        const int l  = lt + m;               // this lane's position
        const float* val = nt ? val1 : val0;
        #pragma unroll
        for (int t = 0; t < 3; ++t) {
            if (lt > id[t] || lt + 15 < pv[t]) continue;   // uniform tile skip
            const bool in = (l >= pv[t]) && (l <= id[t]);
            #pragma unroll
            for (int r = 0; r < 8; ++r) {
                // mask-to-0 safe: ReLU values >= 0 and out initialized to 0
                float mv = in ? val[r] : 0.0f;
                #pragma unroll
                for (int off = 1; off <= 8; off <<= 1)     // within 16-lane half
                    mv = fmaxf(mv, __shfl_xor(mv, off, 32));
                if (m == 0) {                               // lanes 0 and 16
                    const int c = cgb + r + hh * 8;
                    atomicMax((int*)&out[(size_t)bb * OUTS + t * CT + c],
                              __float_as_int(mv));          // order-safe for >=0
                }
            }
        }
    }
}

extern "C" void kernel_launch(void* const* d_in, const int* in_sizes, int n_in,
                              void* d_out, int out_size, void* d_ws, size_t ws_size,
                              hipStream_t stream) {
    const int*   tok = (const int*)  d_in[0];
    const float* emb = (const float*)d_in[1];
    const float* w3f = (const float*)d_in[2];
    const float* b3  = (const float*)d_in[3];
    const float* w5f = (const float*)d_in[4];
    const float* b5  = (const float*)d_in[5];
    const float* w7f = (const float*)d_in[6];
    const float* b7  = (const float*)d_in[7];
    float* out = (float*)d_out;

    // workspace layout (all f16 regions 16B-aligned by construction)
    char* ws = (char*)d_ws;
    size_t off = 0;
    _Float16* xpad = (_Float16*)(ws + off); off += (size_t)B_ * LPAD * E_ * 2; // 8.5 MB
    _Float16* w3h  = (_Float16*)(ws + off); off += (size_t)C_ * 3 * E_ * 2;
    _Float16* w5h  = (_Float16*)(ws + off); off += (size_t)C_ * 5 * E_ * 2;
    _Float16* w7h  = (_Float16*)(ws + off); off += (size_t)C_ * 7 * E_ * 2;
    int*      bnd  = (int*)     (ws + off);

    cvt_f32_f16<<<(C_ * 3 * E_ + 255) / 256, 256, 0, stream>>>(w3f, w3h, C_ * 3 * E_);
    cvt_f32_f16<<<(C_ * 5 * E_ + 255) / 256, 256, 0, stream>>>(w5f, w5h, C_ * 5 * E_);
    cvt_f32_f16<<<(C_ * 7 * E_ + 255) / 256, 256, 0, stream>>>(w7f, w7h, C_ * 7 * E_);
    gather_embed<<<B_ * LPAD, E_, 0, stream>>>(tok, emb, xpad);
    seg_bounds<<<2, 32, 0, stream>>>(tok, bnd);
    zero_out<<<(B_ * OUTS + 255) / 256, 256, 0, stream>>>(out, B_ * OUTS);

    const dim3 grid(L_ / 32, 4, B_);
    conv_wmma_t<3, 1, 0><<<grid, 128, 0, stream>>>(xpad, w3h, b3, bnd, out);
    conv_wmma_t<5, 2, 1><<<grid, 128, 0, stream>>>(xpad, w5h, b5, bnd, out);
    conv_wmma_t<7, 3, 2><<<grid, 128, 0, stream>>>(xpad, w7h, b7, bnd, out);
}
